// MuonLTBlock_66795331388098
// MI455X (gfx1250) — matrix operand order repaired and checked by reading, not compiled, not run on hardware
//
#include <hip/hip_runtime.h>
#include <math.h>

// ---------------------------------------------------------------------------
// MuonLT block for MI455X (gfx1250, wave32, WMMA, TDM).
// Compute-bound (~330 GFLOP vs ~0.3 GB traffic): all GEMMs via
// v_wmma_f32_16x16x32_bf16 (f32 accum). Tile staging via the Tensor Data
// Mover (TENSORcnt) when exposed, else GLOBAL_LOAD_ASYNC_TO_LDS_B128
// (ASYNCcnt), else sync B128. Flash attention (no materialized scores).
// ---------------------------------------------------------------------------

#define BB 2
#define TT 2048
#define DD 1024
#define HH 16
#define HD 64
#define BT (BB * TT)

#ifndef __has_builtin
#define __has_builtin(x) 0
#endif

#if __has_builtin(__builtin_amdgcn_tensor_load_to_lds) && \
    __has_builtin(__builtin_amdgcn_s_wait_tensorcnt)
#define HAVE_TDM 1
#else
#define HAVE_TDM 0
#endif

#if __has_builtin(__builtin_amdgcn_global_load_async_to_lds_b128)
#define HAVE_ASYNC_LDS 1
#else
#define HAVE_ASYNC_LDS 0
#endif

typedef int          v4i __attribute__((ext_vector_type(4)));
typedef unsigned int v4u __attribute__((ext_vector_type(4)));
typedef int          v8i __attribute__((ext_vector_type(8)));

#if HAVE_ASYNC_LDS
#define ASYNC_LOAD_B128(gp, lp)                                                  \
    __builtin_amdgcn_global_load_async_to_lds_b128(                              \
        (__attribute__((address_space(1))) v4i*)(v4i*)(gp),                      \
        (__attribute__((address_space(3))) v4i*)(v4i*)(lp), 0, 0)
#if __has_builtin(__builtin_amdgcn_s_wait_asynccnt)
#define ASYNC_WAIT0() __builtin_amdgcn_s_wait_asynccnt(0)
#else
#define ASYNC_WAIT0() asm volatile("s_wait_asynccnt 0" ::: "memory")
#endif
#else
#define ASYNC_WAIT0()
#endif

// LDS byte offset of a shared-memory pointer
#define LDS_OFF(p) \
    ((unsigned)(unsigned long long)(__attribute__((address_space(3))) void*)(p))

#if HAVE_TDM
// Tensor DMA: load a 2D bf16 tile (tile0 elems wide, tile1 rows) from global
// (row stride = strideElems) into LDS at ldsByteOff, inserting LDS padding of
// padAmtCode (+1 DWORDs) after every 2^(padIntCode+1) DWORDs (ISA ch.8 D#).
__device__ __forceinline__ void tdm_load_2d(const void* gptr, unsigned ldsByteOff,
                                            int strideElems, int tile0, int tile1,
                                            int padIntCode, int padAmtCode) {
    unsigned long long ga = (unsigned long long)(size_t)gptr;
    v4u g0;
    g0.x = 1u;                                            // count=1, user mode
    g0.y = ldsByteOff;                                    // lds_addr (bytes)
    g0.z = (unsigned)(ga & 0xFFFFFFFFu);                  // global_addr[31:0]
    g0.w = (unsigned)((ga >> 32) & 0x01FFFFFFu) | (2u << 30);  // addr[56:32] | type=2
    v8i g1;
    g1[0] = (int)((1u << 16) | (1u << 20) |               // data_size=2B, pad_enable
                  ((unsigned)padIntCode << 22) | ((unsigned)padAmtCode << 25));
    g1[1] = 0;                                            // barrier addr / dim0 lo16=0
    g1[2] = 0x40;                                         // tensor_dim0 = 1<<22 elems
    g1[3] = (tile0 << 16) | 16;                           // tile_dim0 | tensor_dim1 hi (1<<20)
    g1[4] = tile1;                                        // tile_dim1 (tile_dim2=0)
    g1[5] = strideElems;                                  // tensor_dim0_stride lo32
    g1[6] = 0;
    g1[7] = 0;
    v4i z4 = {0, 0, 0, 0};                                // groups 2/3: unused (2D)
    v8i z8 = {0, 0, 0, 0, 0, 0, 0, 0};
    __builtin_amdgcn_tensor_load_to_lds(g0, g1, z4, z4, z8, 0);
}
#define TDM_WAIT0() __builtin_amdgcn_s_wait_tensorcnt(0)
#endif

typedef __attribute__((ext_vector_type(16))) __bf16 v16bf;
typedef __attribute__((ext_vector_type(8)))  float  v8f;

union FragU {
    unsigned u[8];
    v16bf    v;
};
union U4 {
    uint4          v;
    unsigned short s[8];
};
union U2 {
    uint2          v;
    unsigned short s[4];
};

__device__ __forceinline__ unsigned short f2bf(float f) {
    unsigned u = __builtin_bit_cast(unsigned, f);
    u = (u + 0x7FFFu + ((u >> 16) & 1u)) >> 16;
    return (unsigned short)u;
}

__device__ __forceinline__ v8f wmma_bf16(v16bf a, v16bf b, v8f c) {
    return __builtin_amdgcn_wmma_f32_16x16x32_bf16(false, a, false, b, (short)0, c,
                                                   false, false);
}
__device__ __forceinline__ v8f wmma_bf16_ra(v16bf a, v16bf b, v8f c) {
    return __builtin_amdgcn_wmma_f32_16x16x32_bf16(false, a, false, b, (short)0, c,
                                                   true, false);
}

// A fragment: 16x32 bf16 row-major in LDS (lane half h holds k = 16h..16h+15, K-pairs contiguous).
template <int LDS_STRIDE>
__device__ __forceinline__ v16bf load_frag_a(const unsigned short* As, int baseRow,
                                             int k0, int lane) {
    FragU f;
    int m  = baseRow + (lane & 15);
    int kh = (lane & 16) ? 16 : 0;
    const unsigned short* rp = As + m * LDS_STRIDE + k0 + kh;
#pragma unroll
    for (int r = 0; r < 4; ++r) f.u[r] = *(const unsigned*)(rp + 2 * r);
#pragma unroll
    for (int r = 0; r < 4; ++r) f.u[4 + r] = *(const unsigned*)(rp + 8 + 2 * r);
    return f.v;
}

// B fragment: 32x16 bf16 from B^T LDS tile BsT[n][k] (lane half h holds k = 16h..16h+15, same order as A).
template <int LDS_STRIDE>
__device__ __forceinline__ v16bf load_frag_b(const unsigned short* BsT, int baseCol,
                                             int k0, int lane) {
    FragU f;
    int n  = baseCol + (lane & 15);
    int kh = (lane & 16) ? 16 : 0;
    const unsigned short* rp = BsT + n * LDS_STRIDE + k0 + kh;
#pragma unroll
    for (int r = 0; r < 8; ++r) f.u[r] = *(const unsigned*)(rp + 2 * r);
    return f.v;
}

// ---------------------------------------------------------------------------
// Batched bf16 GEMM, C = A*B. 128x128 tile, K-step 32, double-buffered LDS.
// 256 threads = 8 wave32. A tile: one TDM descriptor per tile (wave 0).
// ---------------------------------------------------------------------------
template <int OUT_BF16>
__global__ __launch_bounds__(256) void gemm_bf16_kernel(
    const unsigned short* __restrict__ A, const unsigned short* __restrict__ B,
    void* __restrict__ Cv, int M, int N, int K, int lda, int ldb, int ldc,
    long long strideA, long long strideB, long long strideC) {
    constexpr int LAS = 40;  // 80B row stride: 64B data + 16B pad (16B-aligned)
    constexpr int LBS = 40;
    __shared__ __align__(16) unsigned short As[2][128 * LAS];
    __shared__ __align__(16) unsigned short BsT[2][128 * LBS];

    const int bz = blockIdx.z;
    A += (long long)bz * strideA;
    B += (long long)bz * strideB;

    const int tid   = threadIdx.x;
    const int lane  = tid & 31;
    const int wave  = tid >> 5;
    const int wm    = wave & 3;
    const int wn    = wave >> 2;
    const int tileM = blockIdx.y * 128;
    const int tileN = blockIdx.x * 128;

    auto loadA = [&](int buf, int k0) {
#if HAVE_TDM
        if (wave == 0)  // TDM is per-wave; one descriptor moves the whole tile
            tdm_load_2d(A + (long long)tileM * lda + k0, LDS_OFF(&As[buf][0]),
                        lda, /*tile0=*/32, /*tile1=*/128,
                        /*padInt: 16 DW*/ 3, /*padAmt: 4 DW*/ 3);
#elif HAVE_ASYNC_LDS
#pragma unroll
        for (int j = 0; j < 2; ++j) {
            int c = tid + j * 256;
            int mrow = c >> 2, part = c & 3;
            ASYNC_LOAD_B128(A + (long long)(tileM + mrow) * lda + k0 + part * 8,
                            &As[buf][mrow * LAS + part * 8]);
        }
#else
#pragma unroll
        for (int j = 0; j < 2; ++j) {
            int c = tid + j * 256;
            int mrow = c >> 2, part = c & 3;
            *(uint4*)&As[buf][mrow * LAS + part * 8] =
                *(const uint4*)&A[(long long)(tileM + mrow) * lda + k0 + part * 8];
        }
#endif
    };
    auto waitA = [&]() {
#if HAVE_TDM
        if (wave == 0) TDM_WAIT0();
#elif HAVE_ASYNC_LDS
        ASYNC_WAIT0();
#endif
    };
    // B tile: wide global read -> regs -> transposed LDS store
    auto loadB = [&](int k0, U4* regs) {
#pragma unroll
        for (int j = 0; j < 2; ++j) {
            int c = tid + j * 256;
            int k = c >> 4, nc = (c & 15) << 3;
            regs[j].v = *(const uint4*)&B[(long long)(k0 + k) * ldb + tileN + nc];
        }
    };
    auto storeB = [&](int buf, const U4* regs) {
#pragma unroll
        for (int j = 0; j < 2; ++j) {
            int c = tid + j * 256;
            int k = c >> 4, nc = (c & 15) << 3;
#pragma unroll
            for (int e = 0; e < 8; ++e) BsT[buf][(nc + e) * LBS + k] = regs[j].s[e];
        }
    };

    v8f acc[2][4] = {};
    U4  bregs[2];

    const int nk = K >> 5;
    loadA(0, 0);
    loadB(0, bregs);
    storeB(0, bregs);
    waitA();
    __syncthreads();

    for (int kt = 0; kt < nk; ++kt) {
        const int cur = kt & 1, nxt = cur ^ 1;
        if (kt + 1 < nk) {
            loadA(nxt, (kt + 1) << 5);
            loadB((kt + 1) << 5, bregs);
            if (kt + 2 < nk)
                __builtin_prefetch(&B[(long long)(((kt + 2) << 5) + (tid >> 3)) * ldb + tileN], 0, 1);
        }

        v16bf af[2], bf[4];
#pragma unroll
        for (int mi = 0; mi < 2; ++mi)
            af[mi] = load_frag_a<LAS>(As[cur], wm * 32 + mi * 16, 0, lane);
#pragma unroll
        for (int ni = 0; ni < 4; ++ni)
            bf[ni] = load_frag_b<LBS>(BsT[cur], wn * 64 + ni * 16, 0, lane);
#pragma unroll
        for (int mi = 0; mi < 2; ++mi) {
            acc[mi][0] = wmma_bf16(af[mi], bf[0], acc[mi][0]);
            acc[mi][1] = wmma_bf16_ra(af[mi], bf[1], acc[mi][1]);
            acc[mi][2] = wmma_bf16_ra(af[mi], bf[2], acc[mi][2]);
            acc[mi][3] = wmma_bf16_ra(af[mi], bf[3], acc[mi][3]);
        }

        if (kt + 1 < nk) {
            storeB(nxt, bregs);
            waitA();
        }
        __syncthreads();
    }

    const int rbase = tileM + wm * 32;
    const int cbase = tileN + wn * 64;
    const int rof   = (lane & 16) ? 8 : 0;
#pragma unroll
    for (int mi = 0; mi < 2; ++mi) {
#pragma unroll
        for (int ni = 0; ni < 4; ++ni) {
            int col = cbase + ni * 16 + (lane & 15);
#pragma unroll
            for (int i = 0; i < 8; ++i) {
                int   row = rbase + mi * 16 + i + rof;
                float v   = acc[mi][ni][i];
                if constexpr (OUT_BF16) {
                    unsigned short* C = (unsigned short*)Cv + (long long)bz * strideC;
                    C[(long long)row * ldc + col] = f2bf(v);
                } else {
                    float* C = (float*)Cv + (long long)bz * strideC;
                    C[(long long)row * ldc + col] = v;
                }
            }
        }
    }
}

// ---------------------------------------------------------------------------
// Flash attention, causal. 128 threads = 4 waves per (batch, head, 64-row
// query block). Q/K tiles staged via TDM/async; online softmax; WMMA.
// ---------------------------------------------------------------------------
__global__ __launch_bounds__(128) void attn_kernel(
    const unsigned short* __restrict__ qkv, unsigned short* __restrict__ out) {
    constexpr int LQS = 72;  // 144B row stride: 128B data + 16B pad
    __shared__ __align__(16) unsigned short Qs[64 * LQS];   // [qrow][d]
    __shared__ __align__(16) unsigned short Ks[64 * LQS];   // [key][d]
    __shared__ __align__(16) unsigned short VsT[64 * LQS];  // [d][key]
    __shared__ __align__(16) unsigned short Pld[64 * LQS];  // [qrow][key]
    __shared__ float Sld[64 * 68];
    __shared__ float arow[64];
    __shared__ float lrow[64];

    const int qb = blockIdx.x, h = blockIdx.y, b = blockIdx.z;
    const int q0   = qb * 64;
    const int tid  = threadIdx.x;
    const int lane = tid & 31;
    const int wave = tid >> 5;

    // Q tile (row stride 3*DD elems in global; pad 4 DW after every 32 DW)
#if HAVE_TDM
    if (wave == 0)
        tdm_load_2d(&qkv[(long long)(b * TT + q0) * (3 * DD) + h * HD], LDS_OFF(&Qs[0]),
                    3 * DD, /*tile0=*/64, /*tile1=*/64, /*padInt: 32 DW*/ 4, 3);
#elif HAVE_ASYNC_LDS
    for (int i = tid; i < 64 * 8; i += 128) {
        int r = i >> 3, dc = (i & 7) << 3;
        ASYNC_LOAD_B128(&qkv[(long long)(b * TT + q0 + r) * (3 * DD) + h * HD + dc],
                        &Qs[r * LQS + dc]);
    }
#else
    for (int i = tid; i < 64 * 8; i += 128) {
        int r = i >> 3, dc = (i & 7) << 3;
        *(uint4*)&Qs[r * LQS + dc] =
            *(const uint4*)&qkv[(long long)(b * TT + q0 + r) * (3 * DD) + h * HD + dc];
    }
#endif

    float mreg = -INFINITY, lreg = 0.f;  // threads 0..63 own row = tid
    v8f   oacc[4] = {};
    v8f   zerov   = {};

    for (int kb = 0; kb <= qb; ++kb) {
        const int k0 = kb * 64;
        __syncthreads();
        // K tile
#if HAVE_TDM
        if (wave == 0)
            tdm_load_2d(&qkv[(long long)(b * TT + k0) * (3 * DD) + DD + h * HD],
                        LDS_OFF(&Ks[0]), 3 * DD, 64, 64, 4, 3);
#elif HAVE_ASYNC_LDS
        for (int i = tid; i < 64 * 8; i += 128) {
            int r = i >> 3, dc = (i & 7) << 3;
            ASYNC_LOAD_B128(&qkv[(long long)(b * TT + k0 + r) * (3 * DD) + DD + h * HD + dc],
                            &Ks[r * LQS + dc]);
        }
#else
        for (int i = tid; i < 64 * 8; i += 128) {
            int r = i >> 3, dc = (i & 7) << 3;
            *(uint4*)&Ks[r * LQS + dc] =
                *(const uint4*)&qkv[(long long)(b * TT + k0 + r) * (3 * DD) + DD + h * HD + dc];
        }
#endif
        // V tile, transposed: batch loads to regs, then scatter to LDS
        U2 vr[8];
#pragma unroll
        for (int j = 0; j < 8; ++j) {
            int c = tid + j * 128;
            int r = c >> 4, dg = (c & 15) << 2;
            vr[j].v = *(const uint2*)&qkv[(long long)(b * TT + k0 + r) * (3 * DD) +
                                          2 * DD + h * HD + dg];
        }
#pragma unroll
        for (int j = 0; j < 8; ++j) {
            int c = tid + j * 128;
            int r = c >> 4, dg = (c & 15) << 2;
#pragma unroll
            for (int e = 0; e < 4; ++e) VsT[(dg + e) * LQS + r] = vr[j].s[e];
        }
#if HAVE_TDM
        if (wave == 0) TDM_WAIT0();
#elif HAVE_ASYNC_LDS
        ASYNC_WAIT0();
#endif
        __syncthreads();

        v8f sacc[4];
#pragma unroll
        for (int ni = 0; ni < 4; ++ni) sacc[ni] = zerov;
#pragma unroll
        for (int ks = 0; ks < 64; ks += 32) {
            v16bf aq = load_frag_a<LQS>(Qs, wave * 16, ks, lane);
            v16bf b0 = load_frag_b<LQS>(Ks, 0, ks, lane);
            v16bf b1 = load_frag_b<LQS>(Ks, 16, ks, lane);
            v16bf b2 = load_frag_b<LQS>(Ks, 32, ks, lane);
            v16bf b3 = load_frag_b<LQS>(Ks, 48, ks, lane);
            sacc[0] = wmma_bf16(aq, b0, sacc[0]);
            sacc[1] = wmma_bf16_ra(aq, b1, sacc[1]);
            sacc[2] = wmma_bf16_ra(aq, b2, sacc[2]);
            sacc[3] = wmma_bf16_ra(aq, b3, sacc[3]);
        }
        const int rof = (lane & 16) ? 8 : 0;
#pragma unroll
        for (int ni = 0; ni < 4; ++ni) {
            int coll = ni * 16 + (lane & 15);
#pragma unroll
            for (int i = 0; i < 8; ++i) {
                int   rl = wave * 16 + i + rof;
                float s  = sacc[ni][i] * 0.125f;  // 1/sqrt(HD)
                if (k0 + coll > q0 + rl) s = -1e30f;
                Sld[rl * 68 + coll] = s;
            }
        }
        __syncthreads();

        if (tid < 64) {
            float mx = mreg;
            for (int j = 0; j < 64; ++j) mx = fmaxf(mx, Sld[tid * 68 + j]);
            float alpha = __expf(mreg - mx);
            float s     = 0.f;
            for (int j = 0; j < 64; ++j) {
                float p = __expf(Sld[tid * 68 + j] - mx);
                s += p;
                Pld[tid * LQS + j] = f2bf(p);
            }
            lreg      = lreg * alpha + s;
            mreg      = mx;
            arow[tid] = alpha;
        }
        __syncthreads();

#pragma unroll
        for (int ni = 0; ni < 4; ++ni)
#pragma unroll
            for (int i = 0; i < 8; ++i)
                oacc[ni][i] *= arow[wave * 16 + i + rof];
#pragma unroll
        for (int ks = 0; ks < 64; ks += 32) {
            v16bf ap = load_frag_a<LQS>(Pld, wave * 16, ks, lane);
            v16bf b0 = load_frag_b<LQS>(VsT, 0, ks, lane);
            v16bf b1 = load_frag_b<LQS>(VsT, 16, ks, lane);
            v16bf b2 = load_frag_b<LQS>(VsT, 32, ks, lane);
            v16bf b3 = load_frag_b<LQS>(VsT, 48, ks, lane);
            oacc[0] = wmma_bf16(ap, b0, oacc[0]);
            oacc[1] = wmma_bf16_ra(ap, b1, oacc[1]);
            oacc[2] = wmma_bf16_ra(ap, b2, oacc[2]);
            oacc[3] = wmma_bf16_ra(ap, b3, oacc[3]);
        }
    }

    if (tid < 64) lrow[tid] = lreg;
    __syncthreads();

    const int rof = (lane & 16) ? 8 : 0;
#pragma unroll
    for (int ni = 0; ni < 4; ++ni) {
        int coll = ni * 16 + (lane & 15);
#pragma unroll
        for (int i = 0; i < 8; ++i) {
            int rl = wave * 16 + i + rof;
            out[(long long)(b * TT + q0 + rl) * DD + h * HD + coll] =
                f2bf(oacc[ni][i] / lrow[rl]);
        }
    }
}

// ---------------------------------------------------------------------------
// 32x32 tiled bf16 transpose (per batch), B128 traffic: dst[d][t] = src[t][d].
// ---------------------------------------------------------------------------
__global__ __launch_bounds__(128) void k_transpose_bf(
    const unsigned short* __restrict__ src, unsigned short* __restrict__ dst) {
    __shared__ __align__(16) unsigned short tile[32][40];
    const int b = blockIdx.z;
    const unsigned short* s  = src + (long long)b * TT * DD;
    unsigned short*       dp = dst + (long long)b * TT * DD;
    const int c0 = blockIdx.x * 32;  // d
    const int r0 = blockIdx.y * 32;  // t
    {
        int t = threadIdx.x >> 2, dg = (threadIdx.x & 3) << 3;
        *(uint4*)&tile[t][dg] = *(const uint4*)&s[(long long)(r0 + t) * DD + c0 + dg];
    }
    __syncthreads();
    {
        int dcol = threadIdx.x >> 2, tg = (threadIdx.x & 3) << 3;
        U4 o;
#pragma unroll
        for (int e = 0; e < 8; ++e) o.s[e] = tile[tg + e][dcol];
        *(uint4*)&dp[(long long)(c0 + dcol) * TT + r0 + tg] = o.v;
    }
}

// ---------------------------------------------------------------------------
// Elementwise / reduction kernels
// ---------------------------------------------------------------------------
__global__ void k_scalars(const float* ba, const float* ga, const float* bm,
                          const float* gm, float* s) {
    if (threadIdx.x == 0 && blockIdx.x == 0) {
        s[0] = 1.f / (1.f + __expf(-ba[0]));
        s[1] = log1pf(__expf(ga[0]));
        s[2] = 1.f / (1.f + __expf(-bm[0]));
        s[3] = log1pf(__expf(gm[0]));
    }
}

__global__ __launch_bounds__(256) void k_f2bf(const float* __restrict__ src,
                                              unsigned short* __restrict__ dst,
                                              long long n) {
    long long i = (long long)blockIdx.x * 256 + threadIdx.x;
    if (i < n) dst[i] = f2bf(src[i]);
}

__global__ __launch_bounds__(256) void k_gelu_bf16(const float* __restrict__ src,
                                                   unsigned short* __restrict__ dst,
                                                   long long n) {
    long long i = (long long)blockIdx.x * 256 + threadIdx.x;
    if (i < n) {
        float x = src[i];
        dst[i]  = f2bf(0.5f * x * (1.f + erff(x * 0.70710678118654752f)));
    }
}

__global__ __launch_bounds__(256) void k_lerp(const float* __restrict__ a,
                                              const float* __restrict__ b,
                                              const float* __restrict__ betap,
                                              float* __restrict__ out, long long n) {
    long long i = (long long)blockIdx.x * 256 + threadIdx.x;
    if (i < n) {
        float beta = *betap;
        out[i]     = beta * a[i] + (1.f - beta) * b[i];
    }
}

__global__ __launch_bounds__(256) void k_ln_bf16(const float* __restrict__ x,
                                                 const float* __restrict__ w,
                                                 unsigned short* __restrict__ out) {
    const int row = blockIdx.x;
    const float* xr = x + (long long)row * DD;
    __shared__ float rs[256], rq[256];
    float s = 0.f, q = 0.f;
    for (int i = threadIdx.x; i < DD; i += 256) {
        float v = xr[i];
        s += v;
        q += v * v;
    }
    rs[threadIdx.x] = s;
    rq[threadIdx.x] = q;
    __syncthreads();
    for (int st = 128; st > 0; st >>= 1) {
        if (threadIdx.x < st) {
            rs[threadIdx.x] += rs[threadIdx.x + st];
            rq[threadIdx.x] += rq[threadIdx.x + st];
        }
        __syncthreads();
    }
    float mu  = rs[0] * (1.f / DD);
    float var = rq[0] * (1.f / DD) - mu * mu;
    float inv = rsqrtf(var + 1e-5f);
    for (int i = threadIdx.x; i < DD; i += 256)
        out[(long long)row * DD + i] = f2bf((xr[i] - mu) * inv * w[i]);
}

__global__ __launch_bounds__(256) void k_ln_axpy(const float* __restrict__ base,
                                                 const float* __restrict__ u,
                                                 const float* __restrict__ w,
                                                 const float* __restrict__ gp,
                                                 float* __restrict__ out) {
    const int row = blockIdx.x;
    const float* ur = u + (long long)row * DD;
    __shared__ float rs[256], rq[256];
    float s = 0.f, q = 0.f;
    for (int i = threadIdx.x; i < DD; i += 256) {
        float v = ur[i];
        s += v;
        q += v * v;
    }
    rs[threadIdx.x] = s;
    rq[threadIdx.x] = q;
    __syncthreads();
    for (int st = 128; st > 0; st >>= 1) {
        if (threadIdx.x < st) {
            rs[threadIdx.x] += rs[threadIdx.x + st];
            rq[threadIdx.x] += rq[threadIdx.x + st];
        }
        __syncthreads();
    }
    float mu    = rs[0] * (1.f / DD);
    float var   = rq[0] * (1.f / DD) - mu * mu;
    float inv   = rsqrtf(var + 1e-5f);
    float gamma = *gp;
    for (int i = threadIdx.x; i < DD; i += 256)
        out[(long long)row * DD + i] =
            base[(long long)row * DD + i] + gamma * (ur[i] - mu) * inv * w[i];
}

#define SSQ_BLOCKS 512
__global__ __launch_bounds__(256) void k_sumsq_part(const float* __restrict__ x,
                                                    float* __restrict__ part,
                                                    long long n) {
    const int b = blockIdx.y;
    const float* xb = x + (long long)b * n;
    float s = 0.f;
    for (long long i = (long long)blockIdx.x * 256 + threadIdx.x; i < n;
         i += (long long)gridDim.x * 256) {
        float v = xb[i];
        s += v * v;
    }
    __shared__ float r[256];
    r[threadIdx.x] = s;
    __syncthreads();
    for (int st = 128; st > 0; st >>= 1) {
        if (threadIdx.x < st) r[threadIdx.x] += r[threadIdx.x + st];
        __syncthreads();
    }
    if (threadIdx.x == 0) part[b * SSQ_BLOCKS + blockIdx.x] = r[0];
}

__global__ __launch_bounds__(256) void k_sumsq_final(const float* __restrict__ part,
                                                     float* __restrict__ ss) {
    const int b = blockIdx.x;
    float s = 0.f;
    for (int i = threadIdx.x; i < SSQ_BLOCKS; i += 256) s += part[b * SSQ_BLOCKS + i];
    __shared__ float r[256];
    r[threadIdx.x] = s;
    __syncthreads();
    for (int st = 128; st > 0; st >>= 1) {
        if (threadIdx.x < st) r[threadIdx.x] += r[threadIdx.x + st];
        __syncthreads();
    }
    if (threadIdx.x == 0) ss[b] = r[0];
}

__global__ __launch_bounds__(256) void k_scale_bf(const float* __restrict__ x,
                                                  const float* __restrict__ ss,
                                                  unsigned short* __restrict__ y,
                                                  long long n) {
    const int b = blockIdx.y;
    float rn = 1.f / fmaxf(sqrtf(ss[b]), 1e-12f);
    for (long long i = (long long)blockIdx.x * 256 + threadIdx.x; i < n;
         i += (long long)gridDim.x * 256)
        y[(long long)b * n + i] = f2bf(x[(long long)b * n + i] * rn);
}

__global__ __launch_bounds__(256) void k_nsW(const float* __restrict__ A,
                                             const float* __restrict__ A2,
                                             unsigned short* __restrict__ W,
                                             long long n) {
    long long i = (long long)blockIdx.x * 256 + threadIdx.x;
    if (i < n) {
        int   r = (int)((i / DD) % DD);
        int   c = (int)(i % DD);
        float v = -4.7750f * A[i] + 2.0315f * A2[i];
        if (r == c) v += 3.4445f;
        W[i] = f2bf(v);
    }
}

// ---------------------------------------------------------------------------
// Host orchestration
// ---------------------------------------------------------------------------
extern "C" void kernel_launch(void* const* d_in, const int* in_sizes, int n_in,
                              void* d_out, int out_size, void* d_ws, size_t ws_size,
                              hipStream_t stream) {
    (void)in_sizes; (void)n_in; (void)out_size; (void)ws_size;
    typedef unsigned short u16;
    const float* x    = (const float*)d_in[0];
    const float* m    = (const float*)d_in[1];
    const float* lnA  = (const float*)d_in[2];
    const float* lnM  = (const float*)d_in[3];
    const float* lnUA = (const float*)d_in[4];
    const float* lnUM = (const float*)d_in[5];
    const float* Wqkv = (const float*)d_in[6];
    const float* Wo   = (const float*)d_in[7];
    const float* W1   = (const float*)d_in[8];
    const float* W2   = (const float*)d_in[9];
    const float* bar  = (const float*)d_in[10];
    const float* gar  = (const float*)d_in[11];
    const float* bmr  = (const float*)d_in[12];
    const float* gmr  = (const float*)d_in[13];

    float* x_next = (float*)d_out;
    float* m_next = (float*)d_out + (long long)BT * DD;

    char* wsp = (char*)d_ws;
    auto alloc = [&](size_t bytes) -> char* {
        char* p = wsp;
        wsp += (bytes + 255) & ~(size_t)255;
        return p;
    };
    float* scal    = (float*)alloc(4 * sizeof(float));
    float* ss      = (float*)alloc(2 * sizeof(float));
    float* part    = (float*)alloc(2 * SSQ_BLOCKS * sizeof(float));
    u16*   Wqkv_bf = (u16*)alloc((size_t)DD * 3 * DD * 2);
    u16*   Wo_bf   = (u16*)alloc((size_t)DD * DD * 2);
    u16*   W1_bf   = (u16*)alloc((size_t)DD * 4 * DD * 2);
    u16*   W2_bf   = (u16*)alloc((size_t)4 * DD * DD * 2);
    float* m_half  = (float*)alloc((size_t)BT * DD * 4);
    float* x_half  = (float*)alloc((size_t)BT * DD * 4);
    float* Yf32    = (float*)alloc((size_t)BT * DD * 4);
    u16*   Ybf     = (u16*)alloc((size_t)BT * DD * 2);
    u16*   Ytbf    = (u16*)alloc((size_t)BT * DD * 2);   // Y^T per batch (D x T)
    float* Af32    = (float*)alloc((size_t)BB * DD * DD * 4);
    u16*   Abf     = (u16*)alloc((size_t)BB * DD * DD * 2);
    float* B2f     = (float*)alloc((size_t)BB * DD * DD * 4);
    u16*   Wnsbf   = (u16*)alloc((size_t)BB * DD * DD * 2);
    char*  S       = alloc((size_t)125 * 1024 * 1024);
    // phase-1 layout
    u16*   lnx_bf  = (u16*)(S);
    u16*   qkv_bf  = (u16*)(S + (size_t)8 * 1024 * 1024);
    u16*   atto_bf = (u16*)(S + (size_t)32 * 1024 * 1024);
    float* g1      = (float*)(S + (size_t)40 * 1024 * 1024);
    // phase-2 layout
    u16*   lnm_bf  = (u16*)(S);
    float* hbuf    = (float*)(S + (size_t)8 * 1024 * 1024);
    u16*   h_bf    = (u16*)(S + (size_t)72 * 1024 * 1024);
    float* g2      = (float*)(S + (size_t)104 * 1024 * 1024);

    const long long nBTD = (long long)BT * DD;
    auto eb = [](long long n) { return (unsigned)((n + 255) / 256); };

    k_scalars<<<1, 64, 0, stream>>>(bar, gar, bmr, gmr, scal);
    k_f2bf<<<eb((long long)DD * 3 * DD), 256, 0, stream>>>(Wqkv, Wqkv_bf, (long long)DD * 3 * DD);
    k_f2bf<<<eb((long long)DD * DD), 256, 0, stream>>>(Wo, Wo_bf, (long long)DD * DD);
    k_f2bf<<<eb((long long)DD * 4 * DD), 256, 0, stream>>>(W1, W1_bf, (long long)DD * 4 * DD);
    k_f2bf<<<eb((long long)4 * DD * DD), 256, 0, stream>>>(W2, W2_bf, (long long)4 * DD * DD);

    auto run_ns = [&](const float* Min) {
        k_sumsq_part<<<dim3(SSQ_BLOCKS, BB), 256, 0, stream>>>(Min, part, (long long)TT * DD);
        k_sumsq_final<<<BB, 256, 0, stream>>>(part, ss);
        k_scale_bf<<<dim3(1024, BB), 256, 0, stream>>>(Min, ss, Ybf, (long long)TT * DD);
        for (int it = 0; it < 5; ++it) {
            k_transpose_bf<<<dim3(DD / 32, TT / 32, BB), 128, 0, stream>>>(Ybf, Ytbf);
            // A = Yt @ Y : (D x D), K = T
            gemm_bf16_kernel<0><<<dim3(DD / 128, DD / 128, BB), 256, 0, stream>>>(
                Ytbf, Ybf, Af32, DD, DD, TT, TT, DD, DD,
                (long long)TT * DD, (long long)TT * DD, (long long)DD * DD);
            k_f2bf<<<eb((long long)BB * DD * DD), 256, 0, stream>>>(Af32, Abf,
                                                                    (long long)BB * DD * DD);
            // A2 = A @ A
            gemm_bf16_kernel<0><<<dim3(DD / 128, DD / 128, BB), 256, 0, stream>>>(
                Abf, Abf, B2f, DD, DD, DD, DD, DD, DD,
                (long long)DD * DD, (long long)DD * DD, (long long)DD * DD);
            k_nsW<<<eb((long long)BB * DD * DD), 256, 0, stream>>>(Af32, B2f, Wnsbf,
                                                                   (long long)BB * DD * DD);
            // Y = Y @ W
            gemm_bf16_kernel<0><<<dim3(DD / 128, TT / 128, BB), 256, 0, stream>>>(
                Ybf, Wnsbf, Yf32, TT, DD, DD, DD, DD, DD,
                (long long)TT * DD, (long long)DD * DD, (long long)TT * DD);
            k_f2bf<<<eb(nBTD), 256, 0, stream>>>(Yf32, Ybf, nBTD);
        }
    };

    // ---- attention branch ----
    k_ln_bf16<<<BT, 256, 0, stream>>>(x, lnA, lnx_bf);
    gemm_bf16_kernel<1><<<dim3(3 * DD / 128, BT / 128, 1), 256, 0, stream>>>(
        lnx_bf, Wqkv_bf, qkv_bf, BT, 3 * DD, DD, DD, 3 * DD, 3 * DD, 0, 0, 0);
    attn_kernel<<<dim3(TT / 64, HH, BB), 128, 0, stream>>>(qkv_bf, atto_bf);
    gemm_bf16_kernel<0><<<dim3(DD / 128, BT / 128, 1), 256, 0, stream>>>(
        atto_bf, Wo_bf, g1, BT, DD, DD, DD, DD, DD, 0, 0, 0);
    k_lerp<<<eb(nBTD), 256, 0, stream>>>(m, g1, scal + 0, m_half, nBTD);
    run_ns(m_half);
    k_ln_axpy<<<BT, 256, 0, stream>>>(x, Yf32, lnUA, scal + 1, x_half);

    // ---- MLP branch ----
    k_ln_bf16<<<BT, 256, 0, stream>>>(x_half, lnM, lnm_bf);
    gemm_bf16_kernel<0><<<dim3(4 * DD / 128, BT / 128, 1), 256, 0, stream>>>(
        lnm_bf, W1_bf, hbuf, BT, 4 * DD, DD, DD, 4 * DD, 4 * DD, 0, 0, 0);
    k_gelu_bf16<<<eb((long long)BT * 4 * DD), 256, 0, stream>>>(hbuf, h_bf,
                                                                (long long)BT * 4 * DD);
    gemm_bf16_kernel<0><<<dim3(DD / 128, BT / 128, 1), 256, 0, stream>>>(
        h_bf, W2_bf, g2, BT, DD, 4 * DD, 4 * DD, DD, DD, 0, 0, 0);
    k_lerp<<<eb(nBTD), 256, 0, stream>>>(m_half, g2, scal + 2, m_next, nBTD);
    run_ns(m_next);
    k_ln_axpy<<<BT, 256, 0, stream>>>(x_half, Yf32, lnUM, scal + 3, x_next);
}
